// FeedForwardQuantum_65481071398675
// MI455X (gfx1250) — matrix operand order repaired and checked by reading, not compiled, not run on hardware
//
#include <hip/hip_runtime.h>
#include <hip/hip_bf16.h>

typedef _Float16 v16h __attribute__((ext_vector_type(16)));
typedef float    v8f  __attribute__((ext_vector_type(8)));
typedef _Float16 h2   __attribute__((ext_vector_type(2)));

#define EMBED 768
#define FFN   3072
#define NQ    8
#define MT    128    // tokens per block
#define NT    128    // output cols per block
#define KC    64     // K-chunk (two WMMA K-steps of 32)

// XOR swizzle of the 4-dword block index within a 32-dword (64-half) row
__device__ __forceinline__ int swz(int r) { return (r ^ (r >> 3)) & 7; }

union Frag {
    v16h v;
    unsigned u[8];
    uint4 q[2];
};
union H2x4 {
    h2 h[4];
    uint4 q;
};

__global__ __launch_bounds__(256)
void ffq_fused_kernel(const float* __restrict__ x,
                      const float* __restrict__ theta,
                      const float* __restrict__ w1,
                      const float* __restrict__ b1,
                      const float* __restrict__ w2,
                      const float* __restrict__ b2,
                      float* __restrict__ out)
{
    // swizzled f16 tiles stored as dwords: element (r, kd) at r*32 + ((kd>>2 ^ swz(r))<<2) + (kd&3)
    __shared__ unsigned sHd [MT * 32];      // h chunk    [token][64 halves]  16 KB
    __shared__ unsigned sW2d[NT * 32];      // w2 chunk   [n]    [64 halves]  16 KB
    __shared__ unsigned sW1d[FFN * 4];      // w1 as f16  [k][8 halves]       48 KB
    __shared__ unsigned sQd [MT * 4];       // 0.25*q f16 [tok][8 halves]      2 KB
    __shared__ unsigned sB1d[FFN / 2];      // b1 as packed f16 pairs          6 KB

    const int t    = threadIdx.x;
    const int lane = t & 31;
    const int wave = t >> 5;
    const int row16 = lane & 15;
    const int hi    = lane >> 4;

    const int nt   = blockIdx.x % (EMBED / NT);   // 0..5
    const int mt   = blockIdx.x / (EMBED / NT);   // 0..127
    const int tok0 = mt * MT;
    const int n0   = nt * NT;

    // ---- Phase 0: stage w1 (f16), b1 (packed f16), q' = 0.25*cos(x)*cos(theta) (f16) ----
    for (int idx = t; idx < FFN * 4; idx += 256) {      // w1: pack 2 f32 -> 1 dword
        int k = idx >> 2, p = idx & 3;
        float2 wv = *(const float2*)(w1 + (size_t)k * NQ + 2 * p);
        union { _Float16 h[2]; unsigned u; } pk;
        pk.h[0] = (_Float16)wv.x; pk.h[1] = (_Float16)wv.y;
        sW1d[idx] = pk.u;
    }
    for (int i = t; i < FFN / 2; i += 256) {            // b1 packed pairs
        union { _Float16 h[2]; unsigned u; } pk;
        pk.h[0] = (_Float16)b1[2 * i]; pk.h[1] = (_Float16)b1[2 * i + 1];
        sB1d[i] = pk.u;
    }
    for (int idx = t; idx < MT * 4; idx += 256) {       // q, pre-scaled by 0.25 (4x K-replication)
        int m = idx >> 2, p = idx & 3;
        float2 xv = *(const float2*)(x + (size_t)(tok0 + m) * EMBED + 2 * p);
        float q0 = 0.25f * __builtin_cosf(xv.x) * __builtin_cosf(theta[2 * p]);
        float q1 = 0.25f * __builtin_cosf(xv.y) * __builtin_cosf(theta[2 * p + 1]);
        union { _Float16 h[2]; unsigned u; } pk;
        pk.h[0] = (_Float16)q0; pk.h[1] = (_Float16)q1;
        sQd[idx] = pk.u;
    }
    __syncthreads();

    // ---- persistent q B-fragment for GEMM1, K-replicated (no divergence, no zeros) ----
    // B[K][tok] = 0.25*q[tok][K mod 8] for K=0..31; A will replicate w1 the same way,
    // so the 16x16x32 WMMA computes 4 * 0.25 * (q . w1row) = q . w1row exactly.
    Frag Bq;
    Bq.q[0] = *(const uint4*)(sQd + (wave * 16 + row16) * 4);
    Bq.q[1] = Bq.q[0];

    h2 zero2; zero2[0] = (_Float16)0.0f; zero2[1] = (_Float16)0.0f;

    // ---- GEMM2 accumulators: wave grid 4(M) x 2(N); acc[msub][nsub] ----
    const int wy = wave & 3;     // M strip: rows [32*wy, 32*wy+32)
    const int wx = wave >> 2;    // N half : cols [64*wx, 64*wx+64)
    v8f acc[2][4];
    #pragma unroll
    for (int a = 0; a < 2; a++)
        #pragma unroll
        for (int b = 0; b < 4; b++)
            #pragma unroll
            for (int e = 0; e < 8; e++) acc[a][b][e] = 0.0f;

    const int kq = (t & 15) * 4;   // w2 staging: this thread's k offset (0..60)
    const int nb = t >> 4;         // w2 staging: base n row (0..15)
    const float* wptr = w2 + (size_t)(n0 + nb) * FFN + kq;   // advances KC floats per chunk

    for (int k0 = 0; k0 < FFN; k0 += KC) {
        // ---- GEMM1 via WMMA (transposed, K-replicated): Hsub[k][tok] ----
        #pragma unroll
        for (int ksub = 0; ksub < 4; ksub++) {
            const uint4 wrow = *(const uint4*)(sW1d + (k0 + ksub * 16 + row16) * 4);
            Frag Aw;
            Aw.q[0] = wrow;     // K 0..7 / 8..15  (by lane group)
            Aw.q[1] = wrow;     // K 16..23 / 24..31
            v8f c;
            #pragma unroll
            for (int e = 0; e < 8; e++) c[e] = 0.0f;
            c = __builtin_amdgcn_wmma_f32_16x16x32_f16(
                    false, Aw.v, false, Bq.v, (short)0, c, false, false);
            // lane holds h for token (16*wave+row16), k = k0 + 16*ksub + 8*hi + r (r=0..7)
            H2x4 Bb, R;
            Bb.q = *(const uint4*)(sB1d + (k0 >> 1) + 8 * ksub + 4 * hi);  // b1 pairs
            #pragma unroll
            for (int j = 0; j < 4; j++) {
                h2 p;
                p[0] = (_Float16)c[2 * j];
                p[1] = (_Float16)c[2 * j + 1];      // -> v_cvt_pk_f16_f32
                p = p + Bb.h[j];                    // -> v_pk_add_f16
                R.h[j] = __builtin_elementwise_max(p, zero2);   // -> v_pk_max (ReLU)
            }
            const int mtok = wave * 16 + row16;
            const int bk   = 2 * ksub + hi;                    // 4-dword block in row
            *(uint4*)(sHd + mtok * 32 + ((bk ^ swz(mtok)) << 2)) = R.q;
        }

        // ---- stage w2 chunk as f16 (swizzled); constant instruction offsets ----
        #pragma unroll
        for (int i = 0; i < 8; i++) {
            const int n = nb + 16 * i;
            float4 wv = *(const float4*)(wptr + (size_t)i * 16 * FFN);
            union { _Float16 h[4]; uint2 d; } pk;
            pk.h[0] = (_Float16)wv.x; pk.h[1] = (_Float16)wv.y;
            pk.h[2] = (_Float16)wv.z; pk.h[3] = (_Float16)wv.w;
            *(uint2*)(sW2d + n * 32 + (((kq >> 3) ^ swz(n)) << 2) + ((kq >> 1) & 3)) = pk.d;
        }
        wptr += KC;
        __syncthreads();

        // ---- GEMM2: 2 K-steps x (2 M-sub x 4 N-sub) ----
        #pragma unroll
        for (int ks = 0; ks < 2; ks++) {
            const int m0 = wy * 32 + row16;
            const int m1 = m0 + 16;
            Frag A0, A1;
            {   // A interleaved layout: blocks {ks*4+hi, ks*4+hi+2}
                const int ba = ks * 4 + hi;
                A0.q[0] = *(const uint4*)(sHd + m0 * 32 + (( ba      ^ swz(m0)) << 2));
                A0.q[1] = *(const uint4*)(sHd + m0 * 32 + (((ba + 2) ^ swz(m0)) << 2));
                A1.q[0] = *(const uint4*)(sHd + m1 * 32 + (( ba      ^ swz(m1)) << 2));
                A1.q[1] = *(const uint4*)(sHd + m1 * 32 + (((ba + 2) ^ swz(m1)) << 2));
            }
            #pragma unroll
            for (int ns = 0; ns < 4; ns++) {
                const int nr = wx * 64 + ns * 16 + row16;
                Frag B;   // B interleaved layout: blocks {ks*4+hi, ks*4+hi+2}
                const int bb = ks * 4 + hi;
                B.q[0] = *(const uint4*)(sW2d + nr * 32 + (( bb      ^ swz(nr)) << 2));
                B.q[1] = *(const uint4*)(sW2d + nr * 32 + (((bb + 2) ^ swz(nr)) << 2));
                acc[0][ns] = __builtin_amdgcn_wmma_f32_16x16x32_f16(
                                false, A0.v, false, B.v, (short)0, acc[0][ns], false, false);
                acc[1][ns] = __builtin_amdgcn_wmma_f32_16x16x32_f16(
                                false, A1.v, false, B.v, (short)0, acc[1][ns], false, false);
            }
        }
        __syncthreads();
    }

    // ---- epilogue: out = acc + b2 (C/D layout: VGPR r -> M = r + 8*hi, N = lane%16) ----
    #pragma unroll
    for (int ns = 0; ns < 4; ns++) {
        const int n = n0 + wx * 64 + ns * 16 + row16;
        const float bias = b2[n];
        #pragma unroll
        for (int ms = 0; ms < 2; ms++) {
            #pragma unroll
            for (int r = 0; r < 8; r++) {
                const int m = wy * 32 + ms * 16 + r + 8 * hi;
                out[(size_t)(tok0 + m) * EMBED + n] = acc[ms][ns][r] + bias;
            }
        }
    }
}

extern "C" void kernel_launch(void* const* d_in, const int* in_sizes, int n_in,
                              void* d_out, int out_size, void* d_ws, size_t ws_size,
                              hipStream_t stream) {
    (void)in_sizes; (void)n_in; (void)d_ws; (void)ws_size; (void)out_size;
    const float* x     = (const float*)d_in[0];
    const float* theta = (const float*)d_in[1];
    const float* w1    = (const float*)d_in[2];
    const float* b1    = (const float*)d_in[3];
    const float* w2    = (const float*)d_in[4];
    const float* b2    = (const float*)d_in[5];
    float* out = (float*)d_out;

    const int tokens = 8 * 2048;                 // 16384
    dim3 grid((tokens / MT) * (EMBED / NT));     // 128 * 6 = 768 blocks
    dim3 block(256);                             // 8 wave32
    ffq_fused_kernel<<<grid, block, 0, stream>>>(x, theta, w1, b1, w2, b2, out);
}
